// _QLSTMBase_85478439125726
// MI455X (gfx1250) — compile-verified, matrix operand order checked by reading, not run on hardware
//
#include <hip/hip_runtime.h>

// ---------------------------------------------------------------------------
// QLSTM for MI455X (gfx1250, wave32, WMMA)
//   B=64, T=1024, I=256, H=256, gates 4H=1024
// Phase 0a: quantize weights -> f16 (exact: grid k/16, |k|<=128 fits f16)
// Phase 0b: split x into hi/lo f16 planes (once; ~fp32 accuracy in phase 1)
// Phase 1 : xs = x @ w_ih^T + b via v_wmma_f32_16x16x32_f16
// Phase 2 : persistent 16-WG kernel; WG g owns h columns [16g,16g+16),
//           w_hh slice -> LDS via Tensor Data Mover (padded rows), c in LDS,
//           h exchanged via global double buffer + device barrier per step.
//           A-tile loads batched 4 K-chunks at a time so all global loads of
//           a group are in flight before the first consuming WMMA.
//           Recurrent GEMM is bit-exact (quantized ops exact in f16 + f32 acc).
// ---------------------------------------------------------------------------

typedef __attribute__((ext_vector_type(16))) _Float16 v16h;
typedef __attribute__((ext_vector_type(8)))  _Float16 v8h;
typedef __attribute__((ext_vector_type(8)))  float    v8f;
typedef __attribute__((ext_vector_type(4)))  unsigned v4u32;
typedef __attribute__((ext_vector_type(8)))  int      v8i32;
typedef __attribute__((ext_vector_type(4)))  int      v4i32;

#if defined(__has_builtin)
#if __has_builtin(__builtin_amdgcn_tensor_load_to_lds)
#define HAVE_TDM 1
#endif
#endif

#define B_   64
#define T_   1024
#define I_   256
#define H_   256
#define G4_  1024   // 4*H

// fixed-point fake quant: round-half-even to 1/16 grid, clamp [-8, 7.9375]
__device__ __forceinline__ float fq(float x) {
    float q = rintf(x * 16.0f) * 0.0625f;
    return fminf(fmaxf(q, -8.0f), 7.9375f);
}
__device__ __forceinline__ float sigmoid_f(float x) {
    return 1.0f / (1.0f + __expf(-x));
}
__device__ __forceinline__ float tanh_f(float x) {
    float e = __expf(-2.0f * fabsf(x));
    float t = (1.0f - e) / (1.0f + e);
    return copysignf(t, x);
}

// D = A*B + C, 16x16x32 f16 -> f32
__device__ __forceinline__ v8f wmma_f16(v16h a, v16h b, v8f c) {
    return __builtin_amdgcn_wmma_f32_16x16x32_f16(
        false, a, false, b, (short)0, c, false, false);
}

// Load one 16x32 f16 tile in WMMA A/B layout.
// p must already be the per-lane pointer: base + (row0 + lane%16)*ld + k0 + (lane/16)*8
__device__ __forceinline__ v16h load_tile_f16(const _Float16* p) {
    v8h a = *(const v8h*)p;
    v8h b = *(const v8h*)(p + 16);
    return __builtin_shufflevector(a, b, 0,1,2,3,4,5,6,7,8,9,10,11,12,13,14,15);
}

// ------------------------------ Phase 0a -----------------------------------
__global__ __launch_bounds__(256) void k_prep(
    const float* __restrict__ wih, const float* __restrict__ whh,
    const float* __restrict__ bih, const float* __restrict__ bhh,
    _Float16* __restrict__ wih_q, _Float16* __restrict__ whh_q,
    float* __restrict__ bias, _Float16* __restrict__ hbuf0,
    unsigned* __restrict__ sync) {
    int tid = blockIdx.x * 256 + threadIdx.x;      // 0 .. 262143
    if (tid < G4_ * I_) {
        wih_q[tid] = (_Float16)fq(wih[tid]);
        whh_q[tid] = (_Float16)fq(whh[tid]);
    }
    if (tid < G4_)      bias[tid]  = fq(bih[tid]) + fq(bhh[tid]);
    if (tid < B_ * H_)  hbuf0[tid] = (_Float16)0.0f;
    if (tid == 0)       *sync = 0u;
}

// ------------------------------ Phase 0b -----------------------------------
// x -> (x_hi, x_lo) f16 planes; 8 elements per thread.
__global__ __launch_bounds__(256) void k_xsplit(
    const float* __restrict__ x, _Float16* __restrict__ xhi,
    _Float16* __restrict__ xlo) {
    int i = blockIdx.x * 256 + threadIdx.x;        // 8 floats per thread
    v8f a = *((const v8f*)x + i);
    v8h hi, lo;
#pragma unroll
    for (int j = 0; j < 8; ++j) {
        _Float16 h = (_Float16)a[j];
        hi[j] = h;
        lo[j] = (_Float16)(a[j] - (float)h);
    }
    *((v8h*)xhi + i) = hi;
    *((v8h*)xlo + i) = lo;
}

// ------------------------------- Phase 1 -----------------------------------
// grid = (T_, 16); block = 256 (8 waves). Each WG: 64x64 tile of xs[t*64+b, g].
__global__ __launch_bounds__(256) void k_xproj(
    const _Float16* __restrict__ xhi,   // [B,T,I] f16 hi plane
    const _Float16* __restrict__ xlo,   // [B,T,I] f16 lo plane
    const _Float16* __restrict__ wih,   // [4H,I] quantized f16
    const float* __restrict__ bias,     // [4H]
    float* __restrict__ xs)             // [T,B,4H]
{
    const int t    = blockIdx.x;
    const int nb   = blockIdx.y;
    const int lane = threadIdx.x & 31;
    const int wave = threadIdx.x >> 5;
    const int ntile = wave & 3;        // which 16 cols within the 64-col slab
    const int mpair = wave >> 2;       // rows 0..31 or 32..63
    const int r  = lane & 15;
    const int kh = (lane >> 4) * 8;

    const int col0 = nb * 64 + ntile * 16;
    const int colw = col0 + r;
    const int b0   = mpair * 32;

    const _Float16* bp = wih + (size_t)(col0 + r) * I_ + kh;
    const size_t row0 = ((size_t)(b0 + r)      * T_ + t) * I_ + kh;
    const size_t row1 = ((size_t)(b0 + 16 + r) * T_ + t) * I_ + kh;
    const _Float16* h0p = xhi + row0;
    const _Float16* l0p = xlo + row0;
    const _Float16* h1p = xhi + row1;
    const _Float16* l1p = xlo + row1;

    v8f acc0, acc1;
    const float bv = bias[colw];
#pragma unroll
    for (int v = 0; v < 8; ++v) { acc0[v] = bv; acc1[v] = bv; }

    // batch the 4 A tiles (hi/lo x 2 row-blocks) of 4 K-chunks at a time so the
    // loads issue as one clause before the consuming WMMAs
#pragma unroll
    for (int half = 0; half < 2; ++half) {
        const int kb = half * 128;
        v16h l0t[4], h0t[4], l1t[4], h1t[4];
#pragma unroll
        for (int c = 0; c < 4; ++c) {
            const int k = kb + 32 * c;
            l0t[c] = load_tile_f16(l0p + k);
            h0t[c] = load_tile_f16(h0p + k);
            l1t[c] = load_tile_f16(l1p + k);
            h1t[c] = load_tile_f16(h1p + k);
        }
#pragma unroll
        for (int c = 0; c < 4; ++c) {
            v16h bt = load_tile_f16(bp + kb + 32 * c);
            acc0 = wmma_f16(l0t[c], bt, acc0);
            acc0 = wmma_f16(h0t[c], bt, acc0);
            acc1 = wmma_f16(l1t[c], bt, acc1);
            acc1 = wmma_f16(h1t[c], bt, acc1);
        }
    }

    const int rbase = (lane >> 4) * 8;
    float* o0 = xs + ((size_t)t * B_ + b0      + rbase) * G4_ + colw;
    float* o1 = xs + ((size_t)t * B_ + b0 + 16 + rbase) * G4_ + colw;
#pragma unroll
    for (int v = 0; v < 8; ++v) {
        o0[(size_t)v * G4_] = acc0[v];
        o1[(size_t)v * G4_] = acc1[v];
    }
}

// ------------------------------- Phase 2 -----------------------------------
// grid = 16 persistent WGs, block = 256 (8 waves). WG g owns h cols [16g,16g+16).
__global__ __launch_bounds__(256) void k_lstm(
    const float* __restrict__ xs,        // [T,B,4H]
    const _Float16* __restrict__ whh_q,  // [4H,H] quantized f16
    float* __restrict__ out,             // [B,T,H]
    float* __restrict__ hcT,             // hT then cT, each [B,H]
    _Float16* __restrict__ hbuf,         // [2][B*H] quantized h (f16)
    unsigned* __restrict__ sync)
{
    constexpr int WLD = I_ + 8;          // padded LDS row stride (halfs): 512B + 16B
    constexpr int GLD = 19;              // padded gate row stride (floats)
    __shared__ alignas(16) _Float16 s_whh[64 * WLD];    // 33.8 KB
    __shared__ float s_g[4 * 64 * GLD];                 // 19.5 KB
    __shared__ float s_c[B_ * 16];                      // 4 KB (persistent c state)

    const int wg    = blockIdx.x;        // 0..15
    const int tid   = threadIdx.x;
    const int lane  = tid & 31;
    const int wave  = tid >> 5;
    const int gate  = wave & 3;          // i/f/g/o
    const int mpair = wave >> 2;         // rows 0..31 or 32..63
    const int r  = lane & 15;
    const int kh = (lane >> 4) * 8;
    const int b0 = mpair * 32;

#if defined(HAVE_TDM)
    // Preload the WG's 64 w_hh rows (4 gate blocks of 16 rows x 512B) into LDS
    // with the Tensor Data Mover. LDS padding (4 DWORDs every 128 DWORDs)
    // reproduces the WLD=I_+8 conflict-free stride in hardware.
    if (wave == 0) {
        const unsigned lds_base =
            (unsigned)(unsigned long long)(const void*)&s_whh[0];
#pragma unroll
        for (int q = 0; q < 4; ++q) {
            unsigned long long ga = (unsigned long long)(const void*)
                (whh_q + (size_t)(q * H_ + wg * 16) * I_);
            v4u32 g0 = {
                1u,                                            // count=1 (valid)
                lds_base + (unsigned)(q * 16 * WLD * 2),       // lds_addr
                (unsigned)ga,                                  // global_addr[31:0]
                ((unsigned)(ga >> 32) & 0x01FFFFFFu) | (2u << 30) // [56:32] | type=2
            };
            v8i32 g1 = {
                (int)((2u << 16) | (1u << 20) | (6u << 22) | (3u << 25)),
                //     4B elems  | pad_en    | every 128DW | pad 4 DW
                (int)(128u << 16),   // tensor_dim0 = 128 elements (512B row)
                (int)(16u  << 16),   // tensor_dim0 hi=0 | tensor_dim1 = 16
                (int)(128u << 16),   // tensor_dim1 hi=0 | tile_dim0 = 128
                16,                  // tile_dim1 = 16 rows
                128,                 // tensor_dim0_stride = 128 elements
                0, 0
            };
            v4i32 z4 = {0, 0, 0, 0};
#if __clang_major__ >= 23
            v8i32 z8 = {0, 0, 0, 0, 0, 0, 0, 0};
            __builtin_amdgcn_tensor_load_to_lds(g0, g1, z4, z4, z8, 0);
#else
            __builtin_amdgcn_tensor_load_to_lds(g0, g1, z4, z4, 0);
#endif
        }
        __builtin_amdgcn_s_wait_tensorcnt(0);
    }
#else
    for (int i = tid; i < 64 * (I_ / 8); i += 256) {
        int row = i >> 5, ch = i & 31;            // 32 x 8-half chunks per row
        int q = row >> 4, rr = row & 15;
        const v8h* src = (const v8h*)(whh_q +
            (size_t)(q * H_ + wg * 16 + rr) * I_ + ch * 8);
        *(v8h*)(s_whh + (size_t)row * WLD + ch * 8) = *src;
    }
#endif
    for (int i = tid; i < B_ * 16; i += 256) s_c[i] = 0.0f;
    __syncthreads();

    const int gcol = gate * H_ + wg * 16;         // base gate column in [0,4H)
    const int colw = gcol + r;

    for (int t = 0; t < T_; ++t) {
        const _Float16* hcur  = hbuf + (size_t)(t & 1) * (B_ * H_);
        _Float16*       hnext = hbuf + (size_t)((t + 1) & 1) * (B_ * H_);

        // init accumulators from xs (C layout: lane holds col, vgpr holds row)
        v8f acc0, acc1;
        {
            const float* p0 = xs + ((size_t)t * B_ + b0 + (lane >> 4) * 8) * G4_ + colw;
            const float* p1 = p0 + (size_t)16 * G4_;
#pragma unroll
            for (int v = 0; v < 8; ++v) {
                acc0[v] = p0[(size_t)v * G4_];
                acc1[v] = p1[(size_t)v * G4_];
            }
        }
        if (t + 1 < T_) // warm caches for next step's xs slab (global_prefetch_b8)
            __builtin_prefetch(xs + ((size_t)(t + 1) * B_ + b0 + lane) * G4_ + gcol, 0, 1);

        const _Float16* ap0 = hcur + (size_t)(b0 + r) * H_ + kh;
        const _Float16* ap1 = hcur + (size_t)(b0 + 16 + r) * H_ + kh;
        const _Float16* bp  = s_whh + (size_t)(gate * 16 + r) * WLD + kh;

        // Recurrent GEMM on the T=1024 critical path: batch the global A-tile
        // loads of 4 K-chunks into an indexed array so all 16 global_load_b128
        // issue as one clause before the first consuming WMMA (~2 L2 round
        // trips per step instead of 8).
#pragma unroll
        for (int half = 0; half < 2; ++half) {
            const int kb = half * 128;
            v16h a0t[4], a1t[4];
#pragma unroll
            for (int c = 0; c < 4; ++c) {
                a0t[c] = load_tile_f16(ap0 + kb + 32 * c);
                a1t[c] = load_tile_f16(ap1 + kb + 32 * c);
            }
#pragma unroll
            for (int c = 0; c < 4; ++c) {
                v16h bt = load_tile_f16(bp + kb + 32 * c);   // LDS, padded stride
                acc0 = wmma_f16(a0t[c], bt, acc0);
                acc1 = wmma_f16(a1t[c], bt, acc1);
            }
        }

        // scatter gate tiles to LDS so one thread sees i,f,g,o of one element
        {
            float* gp = s_g + ((size_t)gate * 64 + b0 + (lane >> 4) * 8) * GLD + r;
#pragma unroll
            for (int v = 0; v < 8; ++v) {
                gp[(size_t)v * GLD]        = acc0[v];
                gp[(size_t)(16 + v) * GLD] = acc1[v];
            }
        }
        __syncthreads();

        // elementwise LSTM cell update for this WG's 64x16 slice
        for (int e = tid; e < B_ * 16; e += 256) {
            int m = e >> 4, n = e & 15;
            float ig = s_g[((size_t)0 * 64 + m) * GLD + n];
            float fg = s_g[((size_t)1 * 64 + m) * GLD + n];
            float gg = s_g[((size_t)2 * 64 + m) * GLD + n];
            float og = s_g[((size_t)3 * 64 + m) * GLD + n];
            float i = sigmoid_f(ig);
            float f = sigmoid_f(fg);
            float g = tanh_f(gg);
            float o = sigmoid_f(og);
            float c1 = f * fq(s_c[e]) + i * g;
            float h1 = o * tanh_f(c1);
            s_c[e] = c1;
            int hcol = wg * 16 + n;
            out[((size_t)m * T_ + t) * H_ + hcol] = h1;
            hnext[(size_t)m * H_ + hcol] = (_Float16)fq(h1);   // exact in f16
            if (t == T_ - 1) {
                hcT[(size_t)m * H_ + hcol] = h1;                       // hT
                hcT[(size_t)B_ * H_ + (size_t)m * H_ + hcol] = c1;     // cT
            }
        }

        // ---- device-scope step barrier across the 16 persistent WGs ----
        __syncthreads();
        // Fast path if ever launched as a 16-WG cluster (NOP when ClusterID==0).
        if (__builtin_amdgcn_cluster_id_x() != 0)
            __builtin_amdgcn_s_cluster_barrier();
        if (tid == 0) {
            __threadfence();
            __hip_atomic_fetch_add(sync, 1u, __ATOMIC_RELEASE, __HIP_MEMORY_SCOPE_AGENT);
            unsigned target = 16u * (unsigned)(t + 1);
            while (__hip_atomic_load(sync, __ATOMIC_ACQUIRE, __HIP_MEMORY_SCOPE_AGENT) < target)
                __builtin_amdgcn_s_sleep(2);
            __threadfence();
        }
        __syncthreads();
    }
}

// ---------------------------------------------------------------------------
extern "C" void kernel_launch(void* const* d_in, const int* in_sizes, int n_in,
                              void* d_out, int out_size, void* d_ws, size_t ws_size,
                              hipStream_t stream) {
    (void)in_sizes; (void)n_in; (void)out_size; (void)ws_size;
    const float* x   = (const float*)d_in[0];
    const float* wih = (const float*)d_in[1];
    const float* whh = (const float*)d_in[2];
    const float* bih = (const float*)d_in[3];
    const float* bhh = (const float*)d_in[4];
    float* out = (float*)d_out;
    float* hcT = out + (size_t)B_ * T_ * H_;

    char* ws = (char*)d_ws;
    _Float16* wih_q = (_Float16*)ws;  ws += (size_t)G4_ * I_ * 2;       // 512 KB
    _Float16* whh_q = (_Float16*)ws;  ws += (size_t)G4_ * H_ * 2;       // 512 KB
    float*    bias  = (float*)ws;     ws += (size_t)G4_ * 4;            // 4 KB
    _Float16* hbuf  = (_Float16*)ws;  ws += (size_t)2 * B_ * H_ * 2;    // 64 KB
    unsigned* sync  = (unsigned*)ws;  ws += 256;
    _Float16* xhi   = (_Float16*)ws;  ws += (size_t)B_ * T_ * I_ * 2;   // 33.5 MB
    _Float16* xlo   = (_Float16*)ws;  ws += (size_t)B_ * T_ * I_ * 2;   // 33.5 MB
    float*    xs    = (float*)ws;                                       // 256 MB

    k_prep  <<<dim3((G4_ * I_) / 256), 256, 0, stream>>>(wih, whh, bih, bhh,
                                                         wih_q, whh_q, bias, hbuf, sync);
    k_xsplit<<<dim3((B_ * T_ * I_) / (8 * 256)), 256, 0, stream>>>(x, xhi, xlo);
    k_xproj <<<dim3(T_, 16), 256, 0, stream>>>(xhi, xlo, wih_q, bias, xs);
    k_lstm  <<<dim3(16), 256, 0, stream>>>(xs, whh_q, out, hcT, hbuf, sync);
}
